// TransitionUp_11433202942403
// MI455X (gfx1250) — matrix operand order repaired and checked by reading, not compiled, not run on hardware
//
#include <hip/hip_runtime.h>

// CDNA5 / gfx1250, wave32. WMMA fragment types:
typedef __attribute__((ext_vector_type(16))) _Float16 v16h;  // A/B: 16x32 / 32x16 f16
typedef __attribute__((ext_vector_type(8)))  float    v8f;   // C/D: 16x16 f32

namespace {
constexpr int   kB    = 4;
constexpr int   kN    = 2048;   // coarse points
constexpr int   kM    = 8192;   // fine points
constexpr int   kCin  = 512;
constexpr int   kCout = 256;
constexpr float kBnEps = 1e-5f;
constexpr float kNnEps = 1e-8f;
}

// ---------------------------------------------------------------------------
// Pack activations X (K x Np, fp32 row-major per batch) into B-matrix WMMA
// fragments (f16). Fragment = 32x16 tile (K x N), 512 halves, one v16h/lane.
// Per ISA 7.12.2 (16-bit B): lane l holds column n = l&15; halves h=0..7
// hold k = kb..kb+7 and h=8..15 hold k = kb+16..kb+23, kb = kt*32 + (l&16?8:0).
// ---------------------------------------------------------------------------
__global__ void pack_act_f16(const float* __restrict__ x, _Float16* __restrict__ xp,
                             int K, int Np, int totalFrags) {
  const int lane = threadIdx.x & 31;
  const int frag = blockIdx.x * (blockDim.x >> 5) + (threadIdx.x >> 5);
  if (frag >= totalFrags) return;
  const int KT   = K >> 5;
  const int NT   = Np >> 4;
  const int perB = KT * NT;
  const int b  = frag / perB;
  const int r  = frag - b * perB;
  const int kt = r / NT;
  const int nt = r - kt * NT;
  const int n  = nt * 16 + (lane & 15);
  const int kb = kt * 32 + ((lane & 16) ? 8 : 0);
  const float* src = x + ((size_t)b * K + kb) * Np + n;
  v16h v;
#pragma unroll
  for (int h = 0; h < 16; ++h) v[h] = (_Float16)src[(size_t)(h + (h & 8)) * Np];
  *(v16h*)(xp + ((size_t)frag * 32 + lane) * 16) = v;
}

// ---------------------------------------------------------------------------
// Pack weights W (O x K, fp32 row-major) into A-matrix WMMA fragments (f16).
// Per ISA 7.12.2 (16-bit A 16x32): lane l holds row m = l&15; halves h=0..7
// hold k = kb..kb+7 and h=8..15 hold k = kb+16..kb+23, kb = kt*32 + (l&16?8:0).
// ---------------------------------------------------------------------------
__global__ void pack_wgt_f16(const float* __restrict__ w, _Float16* __restrict__ wp,
                             int K, int totalFrags) {
  const int lane = threadIdx.x & 31;
  const int frag = blockIdx.x * (blockDim.x >> 5) + (threadIdx.x >> 5);
  if (frag >= totalFrags) return;
  const int KT = K >> 5;
  const int ot = frag / KT;
  const int kt = frag - ot * KT;
  const int o  = ot * 16 + (lane & 15);
  const int kb = kt * 32 + ((lane & 16) ? 8 : 0);
  const float* row = w + (size_t)o * K;
  v16h v;
#pragma unroll
  for (int h = 0; h < 8; ++h) v[h]     = (_Float16)row[kb + h];
#pragma unroll
  for (int h = 0; h < 8; ++h) v[8 + h] = (_Float16)row[kb + 16 + h];
  *(v16h*)(wp + ((size_t)frag * 32 + lane) * 16) = v;
}

// ---------------------------------------------------------------------------
// Up branch: f1[b,o,n] = relu(scale[o]*(W_up @ x1)[o,n] + shift[o]).
// One wave computes a 16(o) x 64(n) strip: 1 A-frag x 4 B-frags per k-step,
// 4 accumulators. All fragment loads use compile-time immediate offsets from
// two per-wave base pointers (k-step strides fit the signed 24-bit ioffset).
// ---------------------------------------------------------------------------
__global__ void gemm_up_bn_relu(const _Float16* __restrict__ Ap,  // W_up frags [OT][KT]
                                const _Float16* __restrict__ Bp,  // x1 frags   [b][KT][NT]
                                const float* __restrict__ g,  const float* __restrict__ bt,
                                const float* __restrict__ mu, const float* __restrict__ va,
                                float* __restrict__ f1) {
  const int lane = threadIdx.x & 31;
  const int wave = blockIdx.x * (blockDim.x >> 5) + (threadIdx.x >> 5);
  constexpr int OT = kCout / 16, NT = kN / 16, KT = kCin / 32;
  constexpr int NW = NT / 4;        // n-strips of 4 tiles
  constexpr int perB = OT * NW;
  const int b = wave / perB;        // wave-uniform guard -> EXEC stays all-ones
  if (b >= kB) return;
  const int r   = wave - b * perB;
  const int ot  = r / NW;
  const int ng  = r - ot * NW;
  const int nt0 = ng * 4;

  const _Float16* aBase = Ap + ((size_t)(ot * KT) * 32 + lane) * 16;
  const _Float16* bBase = Bp + ((size_t)(b * KT * NT + nt0) * 32 + lane) * 16;

  v8f a0 = {}, a1 = {}, a2 = {}, a3 = {};
#pragma unroll
  for (int kt = 0; kt < KT; ++kt) {
    const v16h a  = *(const v16h*)(aBase + kt * 512);
    const v16h b0 = *(const v16h*)(bBase + kt * (NT * 512) + 0 * 512);
    const v16h b1 = *(const v16h*)(bBase + kt * (NT * 512) + 1 * 512);
    const v16h b2 = *(const v16h*)(bBase + kt * (NT * 512) + 2 * 512);
    const v16h b3 = *(const v16h*)(bBase + kt * (NT * 512) + 3 * 512);
    a0 = __builtin_amdgcn_wmma_f32_16x16x32_f16(false, a, false, b0, (short)0, a0, false, false);
    a1 = __builtin_amdgcn_wmma_f32_16x16x32_f16(false, a, false, b1, (short)0, a1, false, false);
    a2 = __builtin_amdgcn_wmma_f32_16x16x32_f16(false, a, false, b2, (short)0, a2, false, false);
    a3 = __builtin_amdgcn_wmma_f32_16x16x32_f16(false, a, false, b3, (short)0, a3, false, false);
  }
  // C/D layout: lane l column n=l&15; VGPR rr -> row rr + (l&16 ? 8 : 0).
  const int nl = lane & 15;
  const int rb = (lane & 16) ? 8 : 0;
#pragma unroll
  for (int rr = 0; rr < 8; ++rr) {
    const int   o  = ot * 16 + rb + rr;
    const float sc = g[o] * rsqrtf(va[o] + kBnEps);
    const float sh = bt[o] - mu[o] * sc;
    float* dst = f1 + ((size_t)b * kCout + o) * kN + nt0 * 16 + nl;
    dst[0]  = fmaxf(fmaf(a0[rr], sc, sh), 0.f);
    dst[16] = fmaxf(fmaf(a1[rr], sc, sh), 0.f);
    dst[32] = fmaxf(fmaf(a2[rr], sc, sh), 0.f);
    dst[48] = fmaxf(fmaf(a3[rr], sc, sh), 0.f);
  }
}

// ---------------------------------------------------------------------------
// three_nn + inverse-distance weights. p1[b] (24KB) staged in LDS; the inner
// loop index j is wave-uniform -> LDS reads are broadcasts (no bank conflicts).
// ---------------------------------------------------------------------------
__global__ void three_nn_weights(const float* __restrict__ p1, const float* __restrict__ p2,
                                 int* __restrict__ idx3, float* __restrict__ w3) {
  __shared__ float sp[kN * 3];
  const int b = blockIdx.y;
  for (int t = threadIdx.x; t < kN * 3; t += blockDim.x)
    sp[t] = p1[(size_t)b * kN * 3 + t];
  __syncthreads();
  const int m = blockIdx.x * blockDim.x + threadIdx.x;
  const float px = p2[((size_t)b * kM + m) * 3 + 0];
  const float py = p2[((size_t)b * kM + m) * 3 + 1];
  const float pz = p2[((size_t)b * kM + m) * 3 + 2];
  float d0 = 3.4e38f, d1 = 3.4e38f, d2 = 3.4e38f;
  int   i0 = 0, i1 = 0, i2 = 0;
  for (int j = 0; j < kN; ++j) {
    const float dx = px - sp[3 * j + 0];
    const float dy = py - sp[3 * j + 1];
    const float dz = pz - sp[3 * j + 2];
    const float d  = fmaf(dx, dx, fmaf(dy, dy, dz * dz));
    if (d < d0)      { d2 = d1; i2 = i1; d1 = d0; i1 = i0; d0 = d; i0 = j; }
    else if (d < d1) { d2 = d1; i2 = i1; d1 = d;  i1 = j; }
    else if (d < d2) { d2 = d;  i2 = j; }
  }
  const float r0 = 1.f / (d0 + kNnEps);
  const float r1 = 1.f / (d1 + kNnEps);
  const float r2 = 1.f / (d2 + kNnEps);
  const float inv = 1.f / (r0 + r1 + r2);
  const size_t o = ((size_t)b * kM + m) * 3;
  idx3[o] = i0; idx3[o + 1] = i1; idx3[o + 2] = i2;
  w3[o] = r0 * inv; w3[o + 1] = r1 * inv; w3[o + 2] = r2 * inv;
}

// ---------------------------------------------------------------------------
// Lateral branch + fused interpolation over a 16(o) x 64(m) strip per wave:
//   out[b,o,m] = relu(bn((W_lat @ x2)[o,m])) + sum_k w[b,m,k]*f1[b,o,idx[b,m,k]]
// Each lane owns 4 columns (one per accumulator); idx/weights loaded once per
// column; f1 gathers are L2-resident (f1 = 8MB << 192MB L2).
// ---------------------------------------------------------------------------
__global__ void gemm_lat_interp(const _Float16* __restrict__ Ap,  // W_lat frags
                                const _Float16* __restrict__ Bp,  // x2 frags
                                const float* __restrict__ g,  const float* __restrict__ bt,
                                const float* __restrict__ mu, const float* __restrict__ va,
                                const float* __restrict__ f1,
                                const int*   __restrict__ idx3,
                                const float* __restrict__ w3,
                                float* __restrict__ out) {
  const int lane = threadIdx.x & 31;
  const int wave = blockIdx.x * (blockDim.x >> 5) + (threadIdx.x >> 5);
  constexpr int OT = kCout / 16, NT = kM / 16, KT = kCout / 32;
  constexpr int NW = NT / 4;
  constexpr int perB = OT * NW;
  const int b = wave / perB;
  if (b >= kB) return;
  const int r   = wave - b * perB;
  const int ot  = r / NW;
  const int ng  = r - ot * NW;
  const int nt0 = ng * 4;

  const _Float16* aBase = Ap + ((size_t)(ot * KT) * 32 + lane) * 16;
  const _Float16* bBase = Bp + ((size_t)(b * KT * NT + nt0) * 32 + lane) * 16;

  v8f a0 = {}, a1 = {}, a2 = {}, a3 = {};
#pragma unroll
  for (int kt = 0; kt < KT; ++kt) {
    const v16h a  = *(const v16h*)(aBase + kt * 512);
    const v16h b0 = *(const v16h*)(bBase + kt * (NT * 512) + 0 * 512);
    const v16h b1 = *(const v16h*)(bBase + kt * (NT * 512) + 1 * 512);
    const v16h b2 = *(const v16h*)(bBase + kt * (NT * 512) + 2 * 512);
    const v16h b3 = *(const v16h*)(bBase + kt * (NT * 512) + 3 * 512);
    a0 = __builtin_amdgcn_wmma_f32_16x16x32_f16(false, a, false, b0, (short)0, a0, false, false);
    a1 = __builtin_amdgcn_wmma_f32_16x16x32_f16(false, a, false, b1, (short)0, a1, false, false);
    a2 = __builtin_amdgcn_wmma_f32_16x16x32_f16(false, a, false, b2, (short)0, a2, false, false);
    a3 = __builtin_amdgcn_wmma_f32_16x16x32_f16(false, a, false, b3, (short)0, a3, false, false);
  }

  const int nl = lane & 15;
  const int rb = (lane & 16) ? 8 : 0;
  // per-column interpolation indices/weights (4 columns per lane)
  int   jj[4][3];
  float ww[4][3];
#pragma unroll
  for (int c = 0; c < 4; ++c) {
    const int    mcol = (nt0 + c) * 16 + nl;
    const size_t mo   = ((size_t)b * kM + mcol) * 3;
    jj[c][0] = idx3[mo]; jj[c][1] = idx3[mo + 1]; jj[c][2] = idx3[mo + 2];
    ww[c][0] = w3[mo];   ww[c][1] = w3[mo + 1];   ww[c][2] = w3[mo + 2];
  }
  const float* f1b = f1 + (size_t)b * kCout * kN;
#pragma unroll
  for (int rr = 0; rr < 8; ++rr) {
    const int   o  = ot * 16 + rb + rr;
    const float sc = g[o] * rsqrtf(va[o] + kBnEps);
    const float sh = bt[o] - mu[o] * sc;
    const float* fo  = f1b + (size_t)o * kN;
    float* dst = out + ((size_t)b * kCout + o) * kM + nt0 * 16 + nl;
    const float f2v0 = fmaxf(fmaf(a0[rr], sc, sh), 0.f);
    const float f2v1 = fmaxf(fmaf(a1[rr], sc, sh), 0.f);
    const float f2v2 = fmaxf(fmaf(a2[rr], sc, sh), 0.f);
    const float f2v3 = fmaxf(fmaf(a3[rr], sc, sh), 0.f);
    dst[0]  = f2v0 + ww[0][0]*fo[jj[0][0]] + ww[0][1]*fo[jj[0][1]] + ww[0][2]*fo[jj[0][2]];
    dst[16] = f2v1 + ww[1][0]*fo[jj[1][0]] + ww[1][1]*fo[jj[1][1]] + ww[1][2]*fo[jj[1][2]];
    dst[32] = f2v2 + ww[2][0]*fo[jj[2][0]] + ww[2][1]*fo[jj[2][1]] + ww[2][2]*fo[jj[2][2]];
    dst[48] = f2v3 + ww[3][0]*fo[jj[3][0]] + ww[3][1]*fo[jj[3][1]] + ww[3][2]*fo[jj[3][2]];
  }
}

__global__ void copy_f32(const float* __restrict__ src, float* __restrict__ dst, int n) {
  const int i = blockIdx.x * blockDim.x + threadIdx.x;
  if (i < n) dst[i] = src[i];
}

// ---------------------------------------------------------------------------
extern "C" void kernel_launch(void* const* d_in, const int* in_sizes, int n_in,
                              void* d_out, int out_size, void* d_ws, size_t ws_size,
                              hipStream_t stream) {
  (void)in_sizes; (void)n_in; (void)out_size; (void)ws_size;
  const float* x1    = (const float*)d_in[0];
  const float* p1    = (const float*)d_in[1];
  const float* x2    = (const float*)d_in[2];
  const float* p2    = (const float*)d_in[3];
  const float* w_up  = (const float*)d_in[4];
  const float* g_up  = (const float*)d_in[5];
  const float* b_up  = (const float*)d_in[6];
  const float* m_up  = (const float*)d_in[7];
  const float* v_up  = (const float*)d_in[8];
  const float* w_lat = (const float*)d_in[9];
  const float* g_lat = (const float*)d_in[10];
  const float* b_lat = (const float*)d_in[11];
  const float* m_lat = (const float*)d_in[12];
  const float* v_lat = (const float*)d_in[13];

  // workspace layout (all sections 32B-aligned by construction)
  char* ws = (char*)d_ws;
  float* f1   = (float*)ws;  ws += (size_t)kB * kCout * kN * sizeof(float);      // 8 MB
  int*   idx3 = (int*)ws;    ws += (size_t)kB * kM * 3 * sizeof(int);
  float* wgt3 = (float*)ws;  ws += (size_t)kB * kM * 3 * sizeof(float);
  _Float16* wupP  = (_Float16*)ws; ws += (size_t)(kCout/16)*(kCin/32) *512*sizeof(_Float16);
  _Float16* wlatP = (_Float16*)ws; ws += (size_t)(kCout/16)*(kCout/32)*512*sizeof(_Float16);
  _Float16* x1P   = (_Float16*)ws; ws += (size_t)kB*(kCin/32) *(kN/16)*512*sizeof(_Float16);
  _Float16* x2P   = (_Float16*)ws;                                               // 16 MB

  { // pack weights (tiny, once per launch)
    int frags = (kCout/16)*(kCin/32);                                  // 256
    pack_wgt_f16<<<(frags + 7) / 8, 256, 0, stream>>>(w_up, wupP, kCin, frags);
  }
  {
    int frags = (kCout/16)*(kCout/32);                                 // 128
    pack_wgt_f16<<<(frags + 7) / 8, 256, 0, stream>>>(w_lat, wlatP, kCout, frags);
  }
  { // pack activations fp32 -> f16 fragment layout (single pass over x1/x2)
    int frags = kB * (kCin/32) * (kN/16);                              // 8192
    pack_act_f16<<<frags / 8, 256, 0, stream>>>(x1, x1P, kCin, kN, frags);
  }
  {
    int frags = kB * (kCout/32) * (kM/16);                             // 16384
    pack_act_f16<<<frags / 8, 256, 0, stream>>>(x2, x2P, kCout, kM, frags);
  }
  { // up branch GEMM + BN + ReLU -> f1 (16x64 strip per wave)
    int waves = kB * (kCout/16) * (kN/64);                             // 2048
    gemm_up_bn_relu<<<waves / 8, 256, 0, stream>>>(wupP, x1P, g_up, b_up, m_up, v_up, f1);
  }
  { // 3-NN + weights
    dim3 grid(kM / 256, kB);
    three_nn_weights<<<grid, 256, 0, stream>>>(p1, p2, idx3, wgt3);
  }
  { // lateral GEMM + BN + ReLU + fused 3-point interpolation -> d_out
    int waves = kB * (kCout/16) * (kM/64);                             // 8192
    gemm_lat_interp<<<waves / 8, 256, 0, stream>>>(wlatP, x2P, g_lat, b_lat, m_lat, v_lat,
                                                   f1, idx3, wgt3, (float*)d_out);
  }
  { // p2 pass-through (second tuple element)
    int n = kB * kM * 3;
    copy_f32<<<(n + 255) / 256, 256, 0, stream>>>(p2, (float*)d_out + (size_t)kB * kCout * kM, n);
  }
}
